// MLAAttention_29721173688518
// MI455X (gfx1250) — matrix operand order repaired and checked by reading, not compiled, not run on hardware
//
#include <hip/hip_runtime.h>
#include <hip/hip_bf16.h>
#include <stdint.h>

// ---------------------------------------------------------------------------
// MLA attention for MI455X (gfx1250), bf16 WMMA path with fp32 accumulation.
// Shapes: B=1, S=2048, D=1024, H=16, HKV=8, HD=64, LAT=256.
// ---------------------------------------------------------------------------

typedef __attribute__((ext_vector_type(16))) __bf16 v16bf;
typedef __attribute__((ext_vector_type(8)))  float  v8f;

// Probe for gfx1250 async global->LDS copy builtin (ASYNCcnt-tracked).
#if defined(__has_builtin)
#  if __has_builtin(__builtin_amdgcn_global_load_async_to_lds_b128)
#    define HAVE_ASYNC_LDS 1
#  endif
#endif
#ifndef HAVE_ASYNC_LDS
#  define HAVE_ASYNC_LDS 0
#endif

#if HAVE_ASYNC_LDS
// Builtin signature (from hipcc diagnostic): param0 = int4 addrspace(1)*,
// i.e. GCC-vector of 4 ints in the global address space; param1 = LDS side.
typedef int v4i_vs __attribute__((vector_size(16)));
typedef __attribute__((address_space(1))) v4i_vs* gptr_v4i;
typedef __attribute__((address_space(3))) v4i_vs* lptr_v4i;
#endif

__device__ __forceinline__ __bf16 f2bf(float f) {
  union { float f; unsigned u; } v; v.f = f;
  unsigned r = (v.u + 0x7FFFu + ((v.u >> 16) & 1u)) >> 16;   // round-to-nearest-even
  unsigned short s = (unsigned short)r;
  __bf16 out; __builtin_memcpy(&out, &s, 2);
  return out;
}

// --------------------------- fp32 -> bf16 cast -----------------------------
__global__ void cast_f32_bf16_kernel(const float* __restrict__ in,
                                     __bf16* __restrict__ out, int n) {
  int i = blockIdx.x * blockDim.x + threadIdx.x;
  if (i < n) out[i] = f2bf(in[i]);
}

// ------------------------------- bf16 GEMM ---------------------------------
// C[M,N] = A[M,K] * B[K,N], A/B bf16 row-major, C fp32 or bf16.
// 128-thread block (4 waves). Block tile: M=64, N=32. Each wave owns a 16x32
// C tile; the 32x32 B tile is staged in LDS once per K-step and shared by all
// 4 waves (double-buffered, one barrier per K-step). Staging uses the gfx1250
// async global->LDS path (ASYNCcnt) when the builtin is available.
// Fragment layouts per CDNA5 ISA 7.12.2 (wave32):
//   A 16x32 bf16: lanes 0-15 row M=lane, K = {0..7,16..23}; lanes 16-31 +8.
//   B 32x16 bf16: lanes 0-15 col N=lane, K = {0..7,16..23}; lanes 16-31 +8.
//   C 16x16 f32:  lanes 0-15 (M=vgpr,N=lane); lanes 16-31 (M=vgpr+8,N=lane-16).
template <bool OUT_BF16>
__global__ __launch_bounds__(128) void gemm_bf16_wmma(
    const __bf16* __restrict__ A, const __bf16* __restrict__ B,
    void* __restrict__ Cp, int M, int N, int K) {
  __shared__ alignas(16) __bf16 lds_b[2][32 * 32];

  const int tid = threadIdx.x;
  const int lane = tid & 31;
  const int wave = tid >> 5;
  const int ln = lane & 15;
  const bool hi = lane >= 16;
  const int tm = blockIdx.y * 64 + wave * 16;
  const int tn = blockIdx.x * 32;
  const int row = tm + ln;

  // Cooperative B-tile staging coordinates: 128 threads x 16B = 2048B = 32x32 bf16.
  const int stg_k = tid >> 2;         // 0..31
  const int stg_c = (tid & 3) * 8;    // 0,8,16,24

  v8f acc0 = {}, acc1 = {};
  for (int k0 = 0; k0 < K; k0 += 32) {
    const int buf = (k0 >> 5) & 1;
    // ---- stage B tile into LDS (async if available) ----------------------
    {
      const __bf16* gsrc = B + (size_t)(k0 + stg_k) * N + tn + stg_c;
      __bf16* ldst = &lds_b[buf][stg_k * 32 + stg_c];
#if HAVE_ASYNC_LDS
      __builtin_amdgcn_global_load_async_to_lds_b128(
          (gptr_v4i)(uintptr_t)gsrc, (lptr_v4i)(uintptr_t)ldst, 0, 0);
#else
      *(uint4*)ldst = *(const uint4*)gsrc;
#endif
    }
    // ---- per-wave A fragment from global ---------------------------------
    v16bf a;
#pragma unroll
    for (int i = 0; i < 16; ++i) {
      int j = i >> 1, p = i & 1;
      int kk = ((j < 4) ? (j * 2 + p) : (16 + (j - 4) * 2 + p)) + (hi ? 8 : 0);
      a[i] = A[(size_t)row * K + k0 + kk];
    }
    if (k0 + 32 < K) __builtin_prefetch(&A[(size_t)row * K + k0 + 32], 0, 3);
#if HAVE_ASYNC_LDS
    asm volatile("s_wait_asynccnt 0x0" ::: "memory");
#endif
    __syncthreads();
    // ---- B fragments from LDS -------------------------------------------
    v16bf b0, b1;
#pragma unroll
    for (int i = 0; i < 16; ++i) {
      int kk = ((i < 8 ? i : i + 8) + (hi ? 8 : 0)) * 32;
      b0[i] = lds_b[buf][kk + ln];
      b1[i] = lds_b[buf][kk + 16 + ln];
    }
    acc0 = __builtin_amdgcn_wmma_f32_16x16x32_bf16(false, a, false, b0,
                                                   (short)0, acc0, false, false);
    acc1 = __builtin_amdgcn_wmma_f32_16x16x32_bf16(false, a, false, b1,
                                                   (short)0, acc1, false, false);
  }
#pragma unroll
  for (int v = 0; v < 8; ++v) {
    int r = tm + v + (hi ? 8 : 0);
    if (OUT_BF16) {
      ((__bf16*)Cp)[(size_t)r * N + tn + ln] = f2bf(acc0[v]);
      ((__bf16*)Cp)[(size_t)r * N + tn + 16 + ln] = f2bf(acc1[v]);
    } else {
      ((float*)Cp)[(size_t)r * N + tn + ln] = acc0[v];
      ((float*)Cp)[(size_t)r * N + tn + 16 + ln] = acc1[v];
    }
  }
}

// ----------------------------- RoPE + cast ---------------------------------
// in: fp32 [S, 16*64]; out: roped bf16. One thread per (s, head, i<32) pair.
__global__ void rope_cast_kernel(const float* __restrict__ in,
                                 __bf16* __restrict__ out, int S, float base) {
  int idx = blockIdx.x * blockDim.x + threadIdx.x;
  const int nh = 16;
  if (idx >= S * nh * 32) return;
  int i = idx & 31;
  int h = (idx >> 5) & (nh - 1);
  int s = idx >> 9;  // / (32*nh)
  float inv = __powf(base, -(2.0f * (float)i) / 64.0f);
  float sn, cs;
  __sincosf((float)s * inv, &sn, &cs);
  size_t o = (size_t)s * (nh * 64) + h * 64;
  float x1 = in[o + i];
  float x2 = in[o + 32 + i];
  out[o + i]      = f2bf(x1 * cs - x2 * sn);
  out[o + 32 + i] = f2bf(x2 * cs + x1 * sn);
}

// -------------------------- flash attention --------------------------------
// grid (H=16, S/16). One wave owns a 16-query tile for one head.
// QK^T and PV via v_wmma_f32_16x16x32_bf16; online softmax in fp32.
__global__ __launch_bounds__(32) void mla_flash_attn_kernel(
    const __bf16* __restrict__ Q, const __bf16* __restrict__ Kb,
    const __bf16* __restrict__ Vb, __bf16* __restrict__ O) {
  __shared__ __bf16 lds_p[16 * 32];  // P tile staging for layout transpose
  const int lane = threadIdx.x;
  const int ln = lane & 15;
  const bool hi = lane >= 16;
  const int h = blockIdx.x;
  const int q0 = blockIdx.y * 16;
  const size_t stride = 1024;
  const size_t hoff = (size_t)h * 64;
  const float scale = 0.125f;  // 1/sqrt(64)

  // Resident Q fragments for d=[0,32) and d=[32,64)
  v16bf qa0, qa1;
  {
    const size_t qbase = (size_t)(q0 + ln) * stride + hoff;
#pragma unroll
    for (int i = 0; i < 16; ++i) {
      int j = i >> 1, p = i & 1;
      int kk = ((j < 4) ? (j * 2 + p) : (16 + (j - 4) * 2 + p)) + (hi ? 8 : 0);
      qa0[i] = Q[qbase + kk];
      qa1[i] = Q[qbase + 32 + kk];
    }
  }

  v8f oacc[4] = {};  // O tile 16x64 fp32, 4 N-subtiles of 16
  float m_i[8], l_i[8];
#pragma unroll
  for (int v = 0; v < 8; ++v) { m_i[v] = -1e30f; l_i[v] = 0.0f; }

  for (int kb = 0; kb < q0 + 16; kb += 32) {
    // ---- S = Q * K^T (16x32 keys), 4 WMMAs -------------------------------
    v8f s0 = {}, s1 = {};
    {
      const size_t kb0 = (size_t)(kb + ln) * stride + hoff;
      const size_t kb1 = (size_t)(kb + 16 + ln) * stride + hoff;
      v16bf f0, f1, g0, g1;  // B fragments: K^T tiles, d = {0..7,16..23} (+8 hi lanes)
#pragma unroll
      for (int i = 0; i < 16; ++i) {
        int dd = (i < 8 ? i : i + 8) + (hi ? 8 : 0);
        f0[i] = Kb[kb0 + dd];
        g0[i] = Kb[kb0 + 32 + dd];
        f1[i] = Kb[kb1 + dd];
        g1[i] = Kb[kb1 + 32 + dd];
      }
      s0 = __builtin_amdgcn_wmma_f32_16x16x32_bf16(false, qa0, false, f0,
                                                   (short)0, s0, false, false);
      s0 = __builtin_amdgcn_wmma_f32_16x16x32_bf16(false, qa1, false, g0,
                                                   (short)0, s0, false, false);
      s1 = __builtin_amdgcn_wmma_f32_16x16x32_bf16(false, qa0, false, f1,
                                                   (short)0, s1, false, false);
      s1 = __builtin_amdgcn_wmma_f32_16x16x32_bf16(false, qa1, false, g1,
                                                   (short)0, s1, false, false);
    }

    // ---- causal mask + online softmax (wave32 half-group reductions) -----
    float p0v[8], p1v[8];
#pragma unroll
    for (int v = 0; v < 8; ++v) {
      int qrow = q0 + v + (hi ? 8 : 0);
      float a0 = s0[v] * scale;
      float a1 = s1[v] * scale;
      if (kb + ln > qrow)      a0 = -1e30f;
      if (kb + 16 + ln > qrow) a1 = -1e30f;
      float mx = fmaxf(a0, a1);
#pragma unroll
      for (int d = 1; d < 16; d <<= 1) mx = fmaxf(mx, __shfl_xor(mx, d, 32));
      float mnew = fmaxf(m_i[v], mx);
      float corr = __expf(m_i[v] - mnew);
      m_i[v] = mnew;
      float p0 = __expf(a0 - mnew);
      float p1 = __expf(a1 - mnew);
      float ps = p0 + p1;
#pragma unroll
      for (int d = 1; d < 16; d <<= 1) ps += __shfl_xor(ps, d, 32);
      l_i[v] = l_i[v] * corr + ps;
#pragma unroll
      for (int t = 0; t < 4; ++t) oacc[t][v] *= corr;
      p0v[v] = p0;
      p1v[v] = p1;
    }

    // ---- P: C/D layout -> A layout via LDS -------------------------------
#pragma unroll
    for (int v = 0; v < 8; ++v) {
      int r = v + (hi ? 8 : 0);
      lds_p[r * 32 + ln]      = f2bf(p0v[v]);
      lds_p[r * 32 + 16 + ln] = f2bf(p1v[v]);
    }
    asm volatile("s_wait_dscnt 0x0" ::: "memory");
    v16bf pa;
#pragma unroll
    for (int i = 0; i < 16; ++i) {
      int j = i >> 1, p = i & 1;
      int kk = ((j < 4) ? (j * 2 + p) : (16 + (j - 4) * 2 + p)) + (hi ? 8 : 0);
      pa[i] = lds_p[ln * 32 + kk];
    }

    // ---- O += P * V (32 keys x 64 dims), 4 WMMAs -------------------------
#pragma unroll
    for (int t = 0; t < 4; ++t) {
      v16bf vf;
#pragma unroll
      for (int i = 0; i < 16; ++i) {
        int kk = (i < 8 ? i : i + 8) + (hi ? 8 : 0);
        vf[i] = Vb[(size_t)(kb + kk) * stride + hoff + t * 16 + ln];
      }
      oacc[t] = __builtin_amdgcn_wmma_f32_16x16x32_bf16(false, pa, false, vf,
                                                        (short)0, oacc[t],
                                                        false, false);
    }
  }

  // ---- finalize: O /= l, store bf16 for final projection GEMM -----------
#pragma unroll
  for (int v = 0; v < 8; ++v) {
    float inv_l = 1.0f / l_i[v];
    size_t r = (size_t)(q0 + v + (hi ? 8 : 0));
#pragma unroll
    for (int t = 0; t < 4; ++t)
      O[r * stride + hoff + t * 16 + ln] = f2bf(oacc[t][v] * inv_l);
  }
}

// ---------------------------------------------------------------------------
extern "C" void kernel_launch(void* const* d_in, const int* in_sizes, int n_in,
                              void* d_out, int out_size, void* d_ws,
                              size_t ws_size, hipStream_t stream) {
  (void)n_in; (void)out_size; (void)ws_size;
  const float* x   = (const float*)d_in[0];
  const float* Wq  = (const float*)d_in[1];
  const float* Wk  = (const float*)d_in[2];
  const float* Wv  = (const float*)d_in[3];
  const float* Wkd = (const float*)d_in[4];
  const float* Wvd = (const float*)d_in[5];
  const float* Wku = (const float*)d_in[6];
  const float* Wvu = (const float*)d_in[7];
  const float* Wo  = (const float*)d_in[8];

  const int S = 2048, D = 1024, QD = 1024, KVD = 512, LAT = 256;

  char* ws = (char*)d_ws;
  size_t off = 0;
  auto take = [&](size_t bytes) -> char* {
    char* p = ws + off;
    off += (bytes + 255) & ~(size_t)255;
    return p;
  };
  __bf16* x_bf    = (__bf16*)take((size_t)S * D * 2);
  __bf16* Wq_bf   = (__bf16*)take((size_t)D * QD * 2);
  __bf16* Wk_bf   = (__bf16*)take((size_t)D * KVD * 2);
  __bf16* Wv_bf   = (__bf16*)take((size_t)D * KVD * 2);
  __bf16* Wkd_bf  = (__bf16*)take((size_t)KVD * LAT * 2);
  __bf16* Wvd_bf  = (__bf16*)take((size_t)KVD * LAT * 2);
  __bf16* Wku_bf  = (__bf16*)take((size_t)LAT * QD * 2);
  __bf16* Wvu_bf  = (__bf16*)take((size_t)LAT * QD * 2);
  __bf16* Wo_bf   = (__bf16*)take((size_t)QD * D * 2);
  float*  q_f32   = (float*)take((size_t)S * QD * 4);
  float*  k_f32   = (float*)take((size_t)S * QD * 4);
  __bf16* kmid_bf = (__bf16*)take((size_t)S * KVD * 2);
  __bf16* vmid_bf = (__bf16*)take((size_t)S * KVD * 2);
  __bf16* klat_bf = (__bf16*)take((size_t)S * LAT * 2);
  __bf16* vlat_bf = (__bf16*)take((size_t)S * LAT * 2);
  __bf16* q_bf    = (__bf16*)take((size_t)S * QD * 2);
  __bf16* k_bf    = (__bf16*)take((size_t)S * QD * 2);
  __bf16* v_bf    = (__bf16*)take((size_t)S * QD * 2);
  __bf16* attn_bf = (__bf16*)take((size_t)S * QD * 2);

  auto cast = [&](const float* src, __bf16* dst, int n) {
    cast_f32_bf16_kernel<<<(n + 255) / 256, 256, 0, stream>>>(src, dst, n);
  };
  cast(x,   x_bf,   in_sizes[0]);
  cast(Wq,  Wq_bf,  in_sizes[1]);
  cast(Wk,  Wk_bf,  in_sizes[2]);
  cast(Wv,  Wv_bf,  in_sizes[3]);
  cast(Wkd, Wkd_bf, in_sizes[4]);
  cast(Wvd, Wvd_bf, in_sizes[5]);
  cast(Wku, Wku_bf, in_sizes[6]);
  cast(Wvu, Wvu_bf, in_sizes[7]);
  cast(Wo,  Wo_bf,  in_sizes[8]);

  auto gemm_f32 = [&](const __bf16* A, const __bf16* B, float* C,
                      int M, int N, int K) {
    gemm_bf16_wmma<false><<<dim3(N / 32, M / 64), 128, 0, stream>>>(
        A, B, (void*)C, M, N, K);
  };
  auto gemm_bf = [&](const __bf16* A, const __bf16* B, __bf16* C,
                     int M, int N, int K) {
    gemm_bf16_wmma<true><<<dim3(N / 32, M / 64), 128, 0, stream>>>(
        A, B, (void*)C, M, N, K);
  };

  // Projections
  gemm_f32(x_bf,    Wq_bf,  q_f32,   S, QD,  D);    // q (pre-RoPE)
  gemm_bf (x_bf,    Wk_bf,  kmid_bf, S, KVD, D);    // x@Wk
  gemm_bf (x_bf,    Wv_bf,  vmid_bf, S, KVD, D);    // x@Wv
  gemm_bf (kmid_bf, Wkd_bf, klat_bf, S, LAT, KVD);  // k latent
  gemm_bf (vmid_bf, Wvd_bf, vlat_bf, S, LAT, KVD);  // v latent
  gemm_f32(klat_bf, Wku_bf, k_f32,   S, QD,  LAT);  // k up (pre-RoPE)
  gemm_bf (vlat_bf, Wvu_bf, v_bf,    S, QD,  LAT);  // v up

  // RoPE + cast to bf16
  int rope_threads = S * 16 * 32;
  rope_cast_kernel<<<(rope_threads + 255) / 256, 256, 0, stream>>>(
      q_f32, q_bf, S, 10000.0f);
  rope_cast_kernel<<<(rope_threads + 255) / 256, 256, 0, stream>>>(
      k_f32, k_bf, S, 10000.0f);

  // Flash attention
  mla_flash_attn_kernel<<<dim3(16, S / 16), 32, 0, stream>>>(
      q_bf, k_bf, v_bf, attn_bf);

  // Output projection -> fp32 result
  gemm_f32(attn_bf, Wo_bf, (float*)d_out, S, D, QD);
}
